// DeepseekV2DecoderLayer_74783970558183
// MI455X (gfx1250) — matrix operand order repaired and checked by reading, not compiled, not run on hardware
//
#include <hip/hip_runtime.h>

// ---------------------------------------------------------------------------
// Types for CDNA5 WMMA (wave32, v_wmma_f32_16x16x32_bf16)
// ---------------------------------------------------------------------------
typedef __bf16 bf16_t;
typedef __attribute__((ext_vector_type(16))) __bf16 v16bf;
typedef __attribute__((ext_vector_type(8)))  __bf16 v8bf;
typedef __attribute__((ext_vector_type(8)))  float  v8f;

union AV { v16bf v; v8bf h[2]; };

// ---------------------------------------------------------------------------
// Problem constants (from the reference)
// ---------------------------------------------------------------------------
constexpr int T_    = 2048;
constexpr int DIM_  = 2048;
constexpr int QL_   = 1536;
constexpr int IFF_  = 8192;
constexpr int H_    = 16;
constexpr int DH_   = 256;
constexpr int HI_   = 32;
constexpr int DI_   = 128;
constexpr int TOPK_ = 256;
constexpr int SC_   = 512;     // T / RATIO
constexpr int G_    = 4;
constexpr int OLORA_= 512;
constexpr float THETA_  = 160000.0f;
constexpr float EPS_    = 1e-6f;
constexpr float SCALE_  = 0.0625f;              // 256^-0.5
constexpr float ISCALE_ = 0.08838834764831845f; // 128^-0.5
constexpr float NEGV    = -1e30f;

__device__ __forceinline__ bf16_t f2bf(float f) {
    unsigned u = __float_as_uint(f);
    unsigned r = u + 0x7FFFu + ((u >> 16) & 1u);
    unsigned short h = (unsigned short)(r >> 16);
    return __builtin_bit_cast(bf16_t, h);
}
__device__ __forceinline__ float bf2f(bf16_t b) {
    unsigned short h = __builtin_bit_cast(unsigned short, b);
    return __uint_as_float(((unsigned)h) << 16);
}

// ---------------------------------------------------------------------------
// Weight conversion: f32 [R][C] -> bf16 transposed [C][R]
// ---------------------------------------------------------------------------
__global__ void transpose_f32_to_bf16(const float* __restrict__ in,
                                      bf16_t* __restrict__ out, int R, int C) {
    __shared__ float tile[32][33];
    int c0 = blockIdx.x * 32, r0 = blockIdx.y * 32;
    int tx = threadIdx.x, ty = threadIdx.y; // 32 x 8
    for (int i = ty; i < 32; i += 8) {
        int r = r0 + i, c = c0 + tx;
        tile[i][tx] = (r < R && c < C) ? in[(size_t)r * C + c] : 0.f;
    }
    __syncthreads();
    for (int i = ty; i < 32; i += 8) {
        int c = c0 + i, r = r0 + tx;
        if (c < C && r < R) out[(size_t)c * R + r] = f2bf(tile[tx][i]);
    }
}

__global__ void convert_f32_to_bf16(const float* __restrict__ in,
                                    bf16_t* __restrict__ out, size_t n) {
    size_t i = (size_t)blockIdx.x * blockDim.x + threadIdx.x;
    if (i < n) out[i] = f2bf(in[i]);
}

// ---------------------------------------------------------------------------
// Big bf16 WMMA GEMM: C[M][N] = A[M][K]bf16 x Bt[N][K]bf16 (+res)
//  - REQUIRES: M % 16 == 0, N % 512 == 0, K % 128 == 0
//  - block = 8 waves sharing one 16-row M tile
//  - A tile staged in LDS via double-buffered GLOBAL_LOAD_ASYNC_TO_LDS_B128
//  - each wave computes 4 adjacent 16x16 N tiles, branch-free inner loop
// ---------------------------------------------------------------------------
constexpr int NT_ = 4;    // N tiles per wave
constexpr int KC_ = 128;  // K chunk staged in LDS

__global__ __launch_bounds__(256) void gemm_bf16_big_kernel(
    const bf16_t* __restrict__ A, int lda,
    const bf16_t* __restrict__ Bt, int ldb,
    float* __restrict__ C, bf16_t* __restrict__ Cbf, int ldc,
    const float* __restrict__ addres, int M, int N, int K)
{
    __shared__ __align__(32) bf16_t Asm[2][16 * KC_];   // 2 x 4 KB double buffer
    const int lane = threadIdx.x, wave = threadIdx.y;
    const int tid = wave * 32 + lane;
    const int tm = blockIdx.y;
    const int tn0 = (blockIdx.x * 8 + wave) * NT_;
    const int row16 = lane & 15, hi = lane >> 4;

    const bf16_t* brow[NT_];
#pragma unroll
    for (int j = 0; j < NT_; ++j)
        brow[j] = Bt + (size_t)((tn0 + j) * 16 + row16) * ldb + hi * 16;

    v8f zero = {};
    v8f acc[NT_];
#pragma unroll
    for (int j = 0; j < NT_; ++j) acc[j] = zero;

    // Each thread owns one 16B chunk of the 16 x 128 A tile:
    //   row = tid >> 4  (16 rows), col = (tid & 15) * 8  (128 halfs)
    const int arow_ = tid >> 4;
    const int acol_ = (tid & 15) * 8;
    const bf16_t* agsrc = A + (size_t)(tm * 16 + arow_) * lda + acol_;
    unsigned ldsoff[2];
    ldsoff[0] = (unsigned)(uintptr_t)&Asm[0][arow_ * KC_ + acol_];
    ldsoff[1] = (unsigned)(uintptr_t)&Asm[1][arow_ * KC_ + acol_];

    // Kick off the first async A-tile copy (global -> LDS, ASYNCcnt-tracked)
    {
        const bf16_t* src = agsrc;
        asm volatile("global_load_async_to_lds_b128 %0, %1, off"
                     :: "v"(ldsoff[0]), "v"(src) : "memory");
    }

    int buf = 0;
    for (int kc = 0; kc < K; kc += KC_) {
        if (kc + KC_ < K) {
            const bf16_t* src = agsrc + kc + KC_;
            asm volatile("global_load_async_to_lds_b128 %0, %1, off"
                         :: "v"(ldsoff[buf ^ 1]), "v"(src) : "memory");
            asm volatile("s_wait_asynccnt 1" ::: "memory");  // chunk kc landed
        } else {
            asm volatile("s_wait_asynccnt 0" ::: "memory");
        }
        __syncthreads();   // current buffer fully resident for all waves

        const bf16_t* afrag = &Asm[buf][row16 * KC_ + hi * 16];
#pragma unroll
        for (int k0 = 0; k0 < KC_; k0 += 32) {
            AV a;
            a.h[0] = *(const v8bf*)(afrag + k0);
            a.h[1] = *(const v8bf*)(afrag + k0 + 8);
#pragma unroll
            for (int j = 0; j < NT_; ++j) {
                v16bf b = *(const v16bf*)(brow[j] + kc + k0);
                acc[j] = __builtin_amdgcn_wmma_f32_16x16x32_bf16(
                    false, a.v, false, b, (short)0, acc[j], false, false);
            }
        }
        __syncthreads();   // done reading buf before it is overwritten
        buf ^= 1;
    }

#pragma unroll
    for (int j = 0; j < NT_; ++j) {
        int col = (tn0 + j) * 16 + row16;
#pragma unroll
        for (int r = 0; r < 8; ++r) {
            int rowm = tm * 16 + r + hi * 8;
            size_t off = (size_t)rowm * ldc + col;
            float v = acc[j][r];
            if (addres) v += addres[off];
            if (C)   C[off]   = v;
            if (Cbf) Cbf[off] = f2bf(v);
        }
    }
}

// ---------------------------------------------------------------------------
// Small bf16 WMMA GEMM (ragged N): one wave per 16x16 tile, direct global
// loads.  Requires M % 16 == 0, N % 16 == 0, K % 32 == 0.
// ---------------------------------------------------------------------------
__global__ __launch_bounds__(256) void gemm_bf16_small_kernel(
    const bf16_t* __restrict__ A, int lda,
    const bf16_t* __restrict__ Bt, int ldb,
    float* __restrict__ C, bf16_t* __restrict__ Cbf, int ldc,
    const float* __restrict__ addres, int M, int N, int K)
{
    int lane = threadIdx.x;
    int wave = threadIdx.y;
    int tm = blockIdx.y;
    int tn = blockIdx.x * 8 + wave;
    if (tn * 16 >= N) return;
    int row16 = lane & 15;
    int hi    = lane >> 4;
    const bf16_t* arow = A  + (size_t)(tm * 16 + row16) * lda + hi * 16;
    const bf16_t* brow = Bt + (size_t)(tn * 16 + row16) * ldb + hi * 16;
    v8f c = {};
    for (int k0 = 0; k0 < K; k0 += 32) {
        AV a;
        a.h[0] = *(const v8bf*)(arow + k0);
        a.h[1] = *(const v8bf*)(arow + k0 + 8);
        v16bf b = *(const v16bf*)(brow + k0);
        c = __builtin_amdgcn_wmma_f32_16x16x32_bf16(false, a.v, false, b,
                                                    (short)0, c, false, false);
    }
    int col = tn * 16 + row16;
    for (int r = 0; r < 8; ++r) {
        int rowm = tm * 16 + r + hi * 8;
        size_t off = (size_t)rowm * ldc + col;
        float v = c[r];
        if (addres) v += addres[off];
        if (C)   C[off]   = v;
        if (Cbf) Cbf[off] = f2bf(v);
    }
}

// ---------------------------------------------------------------------------
// Row-wise RMSNorm: out = x * rsqrt(mean(x^2)+eps) * w   (f32/bf16 outputs)
// ---------------------------------------------------------------------------
__global__ void rms_kernel(const float* __restrict__ in, const float* __restrict__ w,
                           float* __restrict__ outf, bf16_t* __restrict__ outb, int D)
{
    int row = blockIdx.x;
    const float* x = in + (size_t)row * D;
    __shared__ float red[256];
    float s = 0.f;
    for (int j = threadIdx.x; j < D; j += blockDim.x) { float v = x[j]; s += v * v; }
    red[threadIdx.x] = s; __syncthreads();
    for (int off = blockDim.x / 2; off > 0; off >>= 1) {
        if ((int)threadIdx.x < off) red[threadIdx.x] += red[threadIdx.x + off];
        __syncthreads();
    }
    float scale = rsqrtf(red[0] / (float)D + EPS_);
    for (int j = threadIdx.x; j < D; j += blockDim.x) {
        float v = x[j] * scale * w[j];
        if (outf) outf[(size_t)row * D + j] = v;
        if (outb) outb[(size_t)row * D + j] = f2bf(v);
    }
}

// ---------------------------------------------------------------------------
// kvg = sigmoid(g_lin)*kv_lin + ape[t%4]
// ---------------------------------------------------------------------------
__global__ void gatekv_kernel(const float* __restrict__ kv, const float* __restrict__ g,
                              const float* __restrict__ ape, float* __restrict__ out, int W)
{
    int t = blockIdx.y;
    int j = blockIdx.x * blockDim.x + threadIdx.x;
    if (j < W) {
        size_t base = (size_t)t * W;
        float gv = 1.f / (1.f + __expf(-g[base + j]));
        out[base + j] = kv[base + j] * gv + ape[(t & 3) * W + j];
    }
}

// ---------------------------------------------------------------------------
// Compression: window mean over 2*RATIO, RMSNorm, RoPE on last 64 dims
// One block per compressed block b; blockDim = d
// ---------------------------------------------------------------------------
__global__ void compress_kernel(const float* __restrict__ kvg,
                                const float* __restrict__ normw,
                                bf16_t* __restrict__ out, int d)
{
    int b = blockIdx.x;
    int j = threadIdx.x;
    __shared__ float row_s[256];
    __shared__ float red[256];
    float acc = 0.f;
    for (int r = 0; r < 4; ++r)
        acc += kvg[(size_t)(b * 4 + r) * (2 * d) + d + j];            // second half, current
    if (b > 0)
        for (int r = 0; r < 4; ++r)
            acc += kvg[(size_t)((b - 1) * 4 + r) * (2 * d) + j];      // first half, previous
    float m = acc * 0.125f;
    red[j] = m * m; __syncthreads();
    for (int off = blockDim.x / 2; off > 0; off >>= 1) {
        if (j < off) red[j] += red[j + off];
        __syncthreads();
    }
    float scale = rsqrtf(red[0] / (float)d + EPS_);
    float v = m * scale * normw[j];
    row_s[j] = v; __syncthreads();
    float outv = v;
    if (j >= d - 64) {
        int k = j - (d - 64);
        int pair = k >> 1;
        float freq = __powf(THETA_, -((float)pair * 2.f / 64.f));
        float ang = (float)(4 * b + 3) * freq;
        float cs = __cosf(ang), sn = __sinf(ang);
        float x0 = row_s[d - 64 + 2 * pair], x1 = row_s[d - 64 + 2 * pair + 1];
        outv = (k & 1) ? (x0 * sn + x1 * cs) : (x0 * cs - x1 * sn);
    }
    out[(size_t)b * d + j] = f2bf(outv);
}

// ---------------------------------------------------------------------------
// RoPE on last 64 dims of each head-vector, f32 -> bf16
// grid = T*Hh, blockDim = Dh
// ---------------------------------------------------------------------------
__global__ void rope_to_bf16_kernel(const float* __restrict__ in,
                                    const int* __restrict__ positions,
                                    bf16_t* __restrict__ out, int Hh, int Dh)
{
    int idx = blockIdx.x;
    int t = idx / Hh, h = idx % Hh;
    size_t base = (size_t)t * Hh * Dh + (size_t)h * Dh;
    int j = threadIdx.x;
    float outv = in[base + j];
    if (j >= Dh - 64) {
        int k = j - (Dh - 64);
        int pair = k >> 1;
        float freq = __powf(THETA_, -((float)pair * 2.f / 64.f));
        float ang = (float)positions[t] * freq;
        float cs = __cosf(ang), sn = __sinf(ang);
        float x0 = in[base + Dh - 64 + 2 * pair], x1 = in[base + Dh - 64 + 2 * pair + 1];
        outv = (k & 1) ? (x0 * sn + x1 * cs) : (x0 * cs - x1 * sn);
    }
    out[base + j] = f2bf(outv);
}

// ---------------------------------------------------------------------------
// Indexer score: iscore[t][s] = sum_h iw[t][h] * relu(qi[t,h,:].ki[s,:]) * ISCALE
// One block (8 waves) per token; WMMA M=32(heads,2 tiles), N=512(Sc), K=128
// ---------------------------------------------------------------------------
__global__ __launch_bounds__(256) void iscore_kernel(
    const bf16_t* __restrict__ qi,   // [T][HI*DI]
    const bf16_t* __restrict__ ki,   // [Sc][DI]  (== Bt layout)
    const float* __restrict__ iw,    // [T][HI]
    float* __restrict__ iscore)      // [T][Sc]
{
    int t = blockIdx.x;
    int lane = threadIdx.x, wave = threadIdx.y;
    int tid = wave * 32 + lane;
    __shared__ float score[SC_];
    __shared__ float iwl[HI_];
    __shared__ __align__(32) bf16_t qis[HI_ * DI_];
    for (int j = tid; j < SC_; j += 256) score[j] = 0.f;
    if (tid < HI_) iwl[tid] = iw[(size_t)t * HI_ + tid];
    for (int j = tid; j < HI_ * DI_; j += 256) qis[j] = qi[(size_t)t * HI_ * DI_ + j];
    __syncthreads();
    int row16 = lane & 15, hi = lane >> 4;
    for (int mt = 0; mt < 2; ++mt) {
        const bf16_t* arow = &qis[(mt * 16 + row16) * DI_ + hi * 16];
        for (int nt = wave; nt < SC_ / 16; nt += 8) {
            const bf16_t* brow = ki + (size_t)(nt * 16 + row16) * DI_ + hi * 16;
            v8f c = {};
            for (int k0 = 0; k0 < DI_; k0 += 32) {
                AV a;
                a.h[0] = *(const v8bf*)(arow + k0);
                a.h[1] = *(const v8bf*)(arow + k0 + 8);
                v16bf b = *(const v16bf*)(brow + k0);
                c = __builtin_amdgcn_wmma_f32_16x16x32_bf16(false, a.v, false, b,
                                                            (short)0, c, false, false);
            }
            float part = 0.f;
            for (int r = 0; r < 8; ++r) {
                int head = mt * 16 + r + hi * 8;
                float v = c[r];
                v = v > 0.f ? v : 0.f;
                part += v * iwl[head];
            }
            atomicAdd(&score[nt * 16 + row16], part * ISCALE_);  // ds_add_f32
        }
    }
    __syncthreads();
    int smax = (t >= 3) ? ((t - 3) >> 2) : -1;   // pos_c[s]=4s+3 <= t
    for (int s = tid; s < SC_; s += 256)
        iscore[(size_t)t * SC_ + s] = (s <= smax) ? score[s] : NEGV;
}

// ---------------------------------------------------------------------------
// Top-K select (K=256 of 512) per token via 32-pass radix select on float keys
// ---------------------------------------------------------------------------
__global__ __launch_bounds__(256) void topk_kernel(const float* __restrict__ iscore,
                                                   int* __restrict__ idxl,
                                                   int* __restrict__ cnt)
{
    int t = blockIdx.x, tid = threadIdx.x;
    __shared__ unsigned keys[SC_];
    __shared__ int redc[256];
    int smax = (t >= 3) ? ((t - 3) >> 2) : -1;
    int nvis = smax + 1; if (nvis > SC_) nvis = SC_;
    for (int s = tid; s < SC_; s += 256) {
        float f = iscore[(size_t)t * SC_ + s];
        unsigned u = __float_as_uint(f);
        u = (u & 0x80000000u) ? ~u : (u | 0x80000000u);
        keys[s] = (s < nvis) ? u : 0u;
    }
    __syncthreads();
    if (nvis <= TOPK_) {
        for (int s = tid; s < nvis; s += 256) idxl[(size_t)t * TOPK_ + s] = s;
        if (tid == 0) cnt[t] = nvis;
        return;
    }
    unsigned pref = 0; int need = TOPK_;
    for (int bit = 31; bit >= 0; --bit) {
        unsigned cand = pref | (1u << bit);
        unsigned maskhi = ~((1u << bit) - 1u);
        int local = 0;
        for (int s = tid; s < SC_; s += 256)
            if ((keys[s] & maskhi) == cand) local++;
        redc[tid] = local; __syncthreads();
        for (int off = 128; off > 0; off >>= 1) {
            if (tid < off) redc[tid] += redc[tid + off];
            __syncthreads();
        }
        int c = redc[0];
        __syncthreads();
        if (c >= need) pref = cand; else need -= c;
    }
    if (tid == 0) {
        int c = 0;
        for (int s = 0; s < nvis && c < TOPK_; ++s)
            if (keys[s] > pref) idxl[(size_t)t * TOPK_ + c++] = s;
        for (int s = 0; s < nvis && c < TOPK_; ++s)
            if (keys[s] == pref) idxl[(size_t)t * TOPK_ + c++] = s;
        cnt[t] = c;
    }
}

// ---------------------------------------------------------------------------
// Sparse attention per token: logits = q.kvc_sel * SCALE, softmax(+sink),
// o = p.kvc_sel.  Selected KV streamed through LDS in 4 chunks of 64 rows.
// ---------------------------------------------------------------------------
__global__ __launch_bounds__(256) void attn_kernel(
    const bf16_t* __restrict__ qb,    // [T][H*DH]
    const bf16_t* __restrict__ kvc,   // [Sc][DH]
    const int* __restrict__ idxl, const int* __restrict__ cnt,
    const float* __restrict__ sink,   // [H]
    bf16_t* __restrict__ ob)          // [T][H*DH]
{
    __shared__ __align__(32) bf16_t kvs[64 * DH_];     // 32 KB
    __shared__ __align__(16) float  logits[H_ * 256];  // 16 KB
    __shared__ __align__(16) bf16_t pmat[H_ * 256];    //  8 KB
    int t = blockIdx.x;
    int lane = threadIdx.x, wave = threadIdx.y;
    int tid = wave * 32 + lane;
    int n = cnt[t];
    int row16 = lane & 15, hi = lane >> 4;
    const bf16_t* qrow = qb + (size_t)t * H_ * DH_ + row16 * DH_ + hi * 16;

    // ---- Phase A: logits over 4 chunks of 64 selected keys ----
    for (int ch = 0; ch < 4; ++ch) {
        for (int e = tid; e < 64 * DH_; e += 256) {
            int r = e >> 8, c2 = e & 255;
            int j = ch * 64 + r;
            kvs[e] = (j < n) ? kvc[(size_t)idxl[(size_t)t * TOPK_ + j] * DH_ + c2]
                             : f2bf(0.f);
        }
        __syncthreads();
        if (wave < 4) {
            int ln = wave * 16 + row16;
            const bf16_t* brow = &kvs[ln * DH_ + hi * 16];
            v8f c = {};
            for (int k0 = 0; k0 < DH_; k0 += 32) {
                AV a;
                a.h[0] = *(const v8bf*)(qrow + k0);
                a.h[1] = *(const v8bf*)(qrow + k0 + 8);
                v16bf b = *(const v16bf*)(brow + k0);
                c = __builtin_amdgcn_wmma_f32_16x16x32_bf16(false, a.v, false, b,
                                                            (short)0, c, false, false);
            }
            int col = ch * 64 + wave * 16 + row16;
            for (int r2 = 0; r2 < 8; ++r2) {
                int h = r2 + hi * 8;
                logits[h * 256 + col] = (col < n) ? c[r2] * SCALE_ : NEGV;
            }
        }
        __syncthreads();
    }

    // ---- Phase B: softmax per head (with sink in denominator) ----
    for (int hh = 0; hh < 2; ++hh) {
        int h = wave * 2 + hh;
        float snk = sink[h];
        float m = snk;
        for (int j = lane; j < 256; j += 32) m = fmaxf(m, logits[h * 256 + j]);
        for (int off = 16; off > 0; off >>= 1) m = fmaxf(m, __shfl_xor(m, off, 32));
        float ssum = 0.f;
        for (int j = lane; j < 256; j += 32) {
            float e_ = __expf(logits[h * 256 + j] - m);
            logits[h * 256 + j] = e_;
            ssum += e_;
        }
        for (int off = 16; off > 0; off >>= 1) ssum += __shfl_xor(ssum, off, 32);
        ssum += __expf(snk - m);
        float inv = 1.f / ssum;
        for (int j = lane; j < 256; j += 32)
            pmat[h * 256 + j] = f2bf(logits[h * 256 + j] * inv);
    }
    __syncthreads();

    // ---- Phase C: o = p @ kvc_sel, streamed over chunks ----
    v8f acc0 = {}, acc1 = {};
    for (int ch = 0; ch < 4; ++ch) {
        for (int e = tid; e < 64 * DH_; e += 256) {
            int r = e >> 8, c2 = e & 255;
            int j = ch * 64 + r;
            kvs[e] = (j < n) ? kvc[(size_t)idxl[(size_t)t * TOPK_ + j] * DH_ + c2]
                             : f2bf(0.f);
        }
        __syncthreads();
        const bf16_t* ar = &pmat[row16 * 256 + ch * 64 + hi * 16];
        for (int half = 0; half < 2; ++half) {
            int dimc = (wave + half * 8) * 16 + row16;
            v8f acc = half ? acc1 : acc0;
            for (int k0 = 0; k0 < 64; k0 += 32) {
                AV a;
                a.h[0] = *(const v8bf*)(ar + k0);
                a.h[1] = *(const v8bf*)(ar + k0 + 8);
                v16bf b;
                int kb = k0 + hi * 16;
                for (int e2 = 0; e2 < 16; ++e2) b[e2] = kvs[(kb + e2) * DH_ + dimc];
                acc = __builtin_amdgcn_wmma_f32_16x16x32_bf16(false, a.v, false, b,
                                                              (short)0, acc, false, false);
            }
            if (half) acc1 = acc; else acc0 = acc;
        }
        __syncthreads();
    }
    for (int half = 0; half < 2; ++half) {
        int dimc = (wave + half * 8) * 16 + row16;
        v8f c = half ? acc1 : acc0;
        for (int r2 = 0; r2 < 8; ++r2) {
            int h = r2 + hi * 8;
            ob[(size_t)t * H_ * DH_ + h * DH_ + dimc] = f2bf(c[r2]);
        }
    }
}

// ---------------------------------------------------------------------------
// mid = silu(gu[:, :I]) * gu[:, I:]   (bf16 in/out)
// ---------------------------------------------------------------------------
__global__ void silu_mul_kernel(const bf16_t* __restrict__ gu,
                                bf16_t* __restrict__ mid)
{
    size_t t = blockIdx.y;
    int j = blockIdx.x * blockDim.x + threadIdx.x;
    if (j < IFF_) {
        float g = bf2f(gu[t * 2 * IFF_ + j]);
        float u = bf2f(gu[t * 2 * IFF_ + IFF_ + j]);
        float s = g / (1.f + __expf(-g));
        mid[t * IFF_ + j] = f2bf(s * u);
    }
}

// ---------------------------------------------------------------------------
// Host orchestration
// ---------------------------------------------------------------------------
extern "C" void kernel_launch(void* const* d_in, const int* in_sizes, int n_in,
                              void* d_out, int out_size, void* d_ws, size_t ws_size,
                              hipStream_t stream)
{
    const float* hidden      = (const float*)d_in[0];
    const int*   positions   = (const int*)  d_in[1];
    const float* ln1_w       = (const float*)d_in[2];
    const float* ln2_w       = (const float*)d_in[3];
    const float* wq_a        = (const float*)d_in[4];
    const float* q_norm_w    = (const float*)d_in[5];
    const float* wq_b        = (const float*)d_in[6];
    const float* comp_wkv    = (const float*)d_in[7];
    const float* comp_wgate  = (const float*)d_in[8];
    const float* comp_ape    = (const float*)d_in[9];
    const float* comp_norm_w = (const float*)d_in[10];
    const float* idx_wq_b    = (const float*)d_in[11];
    const float* idx_wproj   = (const float*)d_in[12];
    const float* icomp_wkv   = (const float*)d_in[13];
    const float* icomp_wgate = (const float*)d_in[14];
    const float* icomp_ape   = (const float*)d_in[15];
    const float* icomp_norm_w= (const float*)d_in[16];
    const float* attn_sink   = (const float*)d_in[17];
    const float* wo_a        = (const float*)d_in[18];
    const float* wo_b        = (const float*)d_in[19];
    const float* gate_up_w   = (const float*)d_in[20];
    const float* down_w      = (const float*)d_in[21];
    float* outp = (float*)d_out;

    size_t cur = 0;
    auto alloc = [&](size_t bytes) -> char* {
        char* p = (char*)d_ws + cur;
        cur += (bytes + 255) & ~(size_t)255;
        return p;
    };

    // bf16 transposed weights [N][K]
    bf16_t* wqaT  = (bf16_t*)alloc((size_t)QL_ * DIM_ * 2);
    bf16_t* wqbT  = (bf16_t*)alloc((size_t)H_ * DH_ * QL_ * 2);
    bf16_t* iwqbT = (bf16_t*)alloc((size_t)HI_ * DI_ * QL_ * 2);
    bf16_t* ckvT  = (bf16_t*)alloc((size_t)2 * DH_ * DIM_ * 2);
    bf16_t* cgT   = (bf16_t*)alloc((size_t)2 * DH_ * DIM_ * 2);
    bf16_t* ikvT  = (bf16_t*)alloc((size_t)2 * DI_ * DIM_ * 2);
    bf16_t* igT   = (bf16_t*)alloc((size_t)2 * DI_ * DIM_ * 2);
    bf16_t* iprojT= (bf16_t*)alloc((size_t)HI_ * DIM_ * 2);
    bf16_t* woaB  = (bf16_t*)alloc((size_t)G_ * OLORA_ * 1024 * 2);
    bf16_t* wobT  = (bf16_t*)alloc((size_t)DIM_ * DIM_ * 2);
    bf16_t* guwT  = (bf16_t*)alloc((size_t)2 * IFF_ * DIM_ * 2);
    bf16_t* dwT   = (bf16_t*)alloc((size_t)DIM_ * IFF_ * 2);

    // activations
    bf16_t* x_bf   = (bf16_t*)alloc((size_t)T_ * DIM_ * 2);
    float*  qpre   = (float*) alloc((size_t)T_ * QL_ * 4);
    bf16_t* qr_bf  = (bf16_t*)alloc((size_t)T_ * QL_ * 2);
    float*  qfull  = (float*) alloc((size_t)T_ * H_ * DH_ * 4);  // reused for qi
    bf16_t* q_bf   = (bf16_t*)alloc((size_t)T_ * H_ * DH_ * 2);
    bf16_t* qi_bf  = (bf16_t*)alloc((size_t)T_ * HI_ * DI_ * 2);
    float*  kvlin  = (float*) alloc((size_t)T_ * 2 * DH_ * 4);
    float*  glin   = (float*) alloc((size_t)T_ * 2 * DH_ * 4);
    float*  kvg    = (float*) alloc((size_t)T_ * 2 * DH_ * 4);
    float*  ikvlin = (float*) alloc((size_t)T_ * 2 * DI_ * 4);
    float*  iglin  = (float*) alloc((size_t)T_ * 2 * DI_ * 4);
    float*  ikvg   = (float*) alloc((size_t)T_ * 2 * DI_ * 4);
    bf16_t* kvc_bf = (bf16_t*)alloc((size_t)SC_ * DH_ * 2);
    bf16_t* ki_bf  = (bf16_t*)alloc((size_t)SC_ * DI_ * 2);
    float*  iw     = (float*) alloc((size_t)T_ * HI_ * 4);
    float*  iscore = (float*) alloc((size_t)T_ * SC_ * 4);
    int*    idxl   = (int*)   alloc((size_t)T_ * TOPK_ * 4);
    int*    cntb   = (int*)   alloc((size_t)T_ * 4);
    bf16_t* o_bf   = (bf16_t*)alloc((size_t)T_ * H_ * DH_ * 2);
    bf16_t* oa_bf  = (bf16_t*)alloc((size_t)T_ * DIM_ * 2);
    float*  hbuf   = (float*) alloc((size_t)T_ * DIM_ * 4);
    bf16_t* x2_bf  = (bf16_t*)alloc((size_t)T_ * DIM_ * 2);
    bf16_t* gu_bf  = (bf16_t*)alloc((size_t)T_ * 2 * IFF_ * 2);
    bf16_t* mid_bf = (bf16_t*)alloc((size_t)T_ * IFF_ * 2);
    (void)ws_size; (void)n_in; (void)in_sizes; (void)out_size;

    dim3 tb32x8(32, 8);
    auto TRS = [&](const float* in, bf16_t* out, int R, int C) {
        transpose_f32_to_bf16<<<dim3((C + 31) / 32, (R + 31) / 32), tb32x8, 0, stream>>>(in, out, R, C);
    };
    auto GEMM = [&](const bf16_t* A, int lda, const bf16_t* Bt, int ldb,
                    float* C, bf16_t* Cb, int ldc, const float* add,
                    int M, int N, int K) {
        if ((N % 512) == 0 && (K % 128) == 0) {
            gemm_bf16_big_kernel<<<dim3(N / 512, M / 16), tb32x8, 0, stream>>>(
                A, lda, Bt, ldb, C, Cb, ldc, add, M, N, K);
        } else {
            gemm_bf16_small_kernel<<<dim3((N + 127) / 128, M / 16), tb32x8, 0, stream>>>(
                A, lda, Bt, ldb, C, Cb, ldc, add, M, N, K);
        }
    };

    // ---- weight conversion ----
    TRS(wq_a,      wqaT,  DIM_, QL_);
    TRS(wq_b,      wqbT,  QL_,  H_ * DH_);
    TRS(idx_wq_b,  iwqbT, QL_,  HI_ * DI_);
    TRS(comp_wkv,  ckvT,  DIM_, 2 * DH_);
    TRS(comp_wgate,cgT,   DIM_, 2 * DH_);
    TRS(icomp_wkv, ikvT,  DIM_, 2 * DI_);
    TRS(icomp_wgate,igT,  DIM_, 2 * DI_);
    TRS(idx_wproj, iprojT,DIM_, HI_);
    TRS(wo_b,      wobT,  DIM_, DIM_);
    TRS(gate_up_w, guwT,  DIM_, 2 * IFF_);
    TRS(down_w,    dwT,   IFF_, DIM_);
    {
        size_t n = (size_t)G_ * OLORA_ * 1024;
        convert_f32_to_bf16<<<(unsigned)((n + 255) / 256), 256, 0, stream>>>(wo_a, woaB, n);
    }

    // ---- pre-attention pipeline ----
    rms_kernel<<<T_, 256, 0, stream>>>(hidden, ln1_w, nullptr, x_bf, DIM_);

    GEMM(x_bf, DIM_, wqaT, DIM_, qpre, nullptr, QL_, nullptr, T_, QL_, DIM_);
    rms_kernel<<<T_, 256, 0, stream>>>(qpre, q_norm_w, nullptr, qr_bf, QL_);

    GEMM(qr_bf, QL_, wqbT, QL_, qfull, nullptr, H_ * DH_, nullptr, T_, H_ * DH_, QL_);
    rope_to_bf16_kernel<<<T_ * H_, DH_, 0, stream>>>(qfull, positions, q_bf, H_, DH_);

    GEMM(qr_bf, QL_, iwqbT, QL_, qfull, nullptr, HI_ * DI_, nullptr, T_, HI_ * DI_, QL_);
    rope_to_bf16_kernel<<<T_ * HI_, DI_, 0, stream>>>(qfull, positions, qi_bf, HI_, DI_);

    // compression (main KV)
    GEMM(x_bf, DIM_, ckvT, DIM_, kvlin, nullptr, 2 * DH_, nullptr, T_, 2 * DH_, DIM_);
    GEMM(x_bf, DIM_, cgT,  DIM_, glin,  nullptr, 2 * DH_, nullptr, T_, 2 * DH_, DIM_);
    gatekv_kernel<<<dim3((2 * DH_ + 255) / 256, T_), 256, 0, stream>>>(kvlin, glin, comp_ape, kvg, 2 * DH_);
    compress_kernel<<<SC_, DH_, 0, stream>>>(kvg, comp_norm_w, kvc_bf, DH_);

    // compression (indexer KV)  (N=256 -> small-GEMM path)
    GEMM(x_bf, DIM_, ikvT, DIM_, ikvlin, nullptr, 2 * DI_, nullptr, T_, 2 * DI_, DIM_);
    GEMM(x_bf, DIM_, igT,  DIM_, iglin,  nullptr, 2 * DI_, nullptr, T_, 2 * DI_, DIM_);
    gatekv_kernel<<<dim3((2 * DI_ + 255) / 256, T_), 256, 0, stream>>>(ikvlin, iglin, icomp_ape, ikvg, 2 * DI_);
    compress_kernel<<<SC_, DI_, 0, stream>>>(ikvg, icomp_norm_w, ki_bf, DI_);

    // indexer weights (N=32 -> small-GEMM path), scores, top-k
    GEMM(x_bf, DIM_, iprojT, DIM_, iw, nullptr, HI_, nullptr, T_, HI_, DIM_);
    iscore_kernel<<<T_, tb32x8, 0, stream>>>(qi_bf, ki_bf, iw, iscore);
    topk_kernel<<<T_, 256, 0, stream>>>(iscore, idxl, cntb);

    // sparse attention
    attn_kernel<<<T_, tb32x8, 0, stream>>>(q_bf, kvc_bf, idxl, cntb, attn_sink, o_bf);

    // output projection: grouped LoRA (wo_a already [N][K] per group), then wo_b
    for (int g = 0; g < G_; ++g) {
        GEMM(o_bf + g * 1024, H_ * DH_,
             woaB + (size_t)g * OLORA_ * 1024, 1024,
             nullptr, oa_bf + g * OLORA_, DIM_, nullptr,
             T_, OLORA_, 1024);
    }
    GEMM(oa_bf, DIM_, wobT, DIM_, hbuf, nullptr, DIM_, hidden, T_, DIM_, DIM_);

    // MLP
    rms_kernel<<<T_, 256, 0, stream>>>(hbuf, ln2_w, nullptr, x2_bf, DIM_);
    GEMM(x2_bf, DIM_, guwT, DIM_, nullptr, gu_bf, 2 * IFF_, nullptr, T_, 2 * IFF_, DIM_);
    silu_mul_kernel<<<dim3(IFF_ / 256, T_), 256, 0, stream>>>(gu_bf, mid_bf);
    GEMM(mid_bf, IFF_, dwT, IFF_, outp, nullptr, DIM_, hbuf, T_, DIM_, IFF_);
}
